// SimplePairformerBlock_33921651704337
// MI455X (gfx1250) — matrix operand order repaired and checked by reading, not compiled, not run on hardware
//
#include <hip/hip_runtime.h>

// ---------------------------------------------------------------------------
// SimplePairformerBlock for MI455X (gfx1250), wave32 + WMMA bf16.
// N=768 tokens, Cz=128, Cs=384.  ~250 GFLOP total, bandwidth-bound at
// 23.3 TB/s -> bf16 operands / f32 accumulation, channel-major re-layout so
// the 128x (768^3) triangle GEMMs use contiguous b128 fragment loads and stay
// L2-resident per channel.  k2 uses 2x2 register blocking (4 WMMA / 8 loads).
// ---------------------------------------------------------------------------

#define N_TOK 768
#define CZ    128
#define CS    384

typedef __bf16   bhalf_t;
typedef bhalf_t  bhalf8  __attribute__((ext_vector_type(8)));
typedef bhalf_t  bhalf16 __attribute__((ext_vector_type(16)));
typedef float    floatx4 __attribute__((ext_vector_type(4)));
typedef float    floatx8 __attribute__((ext_vector_type(8)));

// ---- WMMA wrapper: D = A(16x32 bf16) * B(32x16 bf16) + C(16x16 f32) -------
__device__ __forceinline__ floatx8 wmma32(bhalf16 a, bhalf16 b, floatx8 c) {
  return __builtin_amdgcn_wmma_f32_16x16x32_bf16(
      /*neg_a=*/false, a, /*neg_b=*/false, b,
      /*c_mod=*/(short)0, c, /*reuse_a=*/false, /*reuse_b=*/false);
}

__device__ __forceinline__ bhalf16 ld_pair(const bhalf_t* p0, const bhalf_t* p1) {
  union { bhalf16 v; bhalf8 h[2]; } u;
  u.h[0] = *(const bhalf8*)p0;
  u.h[1] = *(const bhalf8*)p1;
  return u.v;
}

// A-fragment 16x32 (M x K), row-major source with row stride `stride`.
// lane L: M = L&15; elem e -> K = kbase + off + e (e = 0..15)
// where off = (L>=16 ? 16 : 0).
__device__ __forceinline__ bhalf16 frag_a(const bhalf_t* base, int stride,
                                          int kbase, int lane) {
  const bhalf_t* p = base + (lane & 15) * stride + kbase + ((lane & 16) ? 16 : 0);
  return ld_pair(p, p + 8);
}

// B-fragment 32x16 (K x N) sourced from row-major [N][K] storage (i.e. B^T
// rows contiguous over K): lane L: N = L&15; K = kbase + e + (L>=16 ? 16 : 0).
__device__ __forceinline__ bhalf16 frag_bt(const bhalf_t* base, int stride,
                                           int kbase, int lane) {
  const bhalf_t* p = base + (lane & 15) * stride + kbase + ((lane & 16) ? 16 : 0);
  return ld_pair(p, p + 8);
}

// B-fragment from pre-packed weights: frag-major [(kc*ntiles+nt)][lane][16].
__device__ __forceinline__ bhalf16 frag_pk(const bhalf_t* pack, int kc, int nt,
                                           int ntiles, int lane) {
  const bhalf_t* p = pack + (((size_t)kc * ntiles + nt) << 9) + (lane << 4);
  return ld_pair(p, p + 8);
}

// butterfly sum over 16-lane subgroup (stays inside wave32)
__device__ __forceinline__ void red16(float& a, float& b) {
  for (int m = 8; m >= 1; m >>= 1) {
    a += __shfl_xor(a, m, 32);
    b += __shfl_xor(b, m, 32);
  }
}

// ---------------------------------------------------------------------------
__global__ void k_zero(float* p, int n) {
  int i = blockIdx.x * 256 + threadIdx.x;
  if (i < n) p[i] = 0.f;
}

// Pack W[K][Ncols] f32 -> B-fragment-ordered bf16.
__global__ void k_pack_b(const float* __restrict__ W, bhalf_t* __restrict__ dst,
                         int K, int Ncols) {
  int idx = blockIdx.x * 256 + threadIdx.x;
  if (idx >= K * Ncols) return;
  int ntiles = Ncols >> 4;
  int f = idx >> 9, r = idx & 511;
  int lane = r >> 4, e = r & 15;
  int kc = f / ntiles, nt = f % ntiles;
  int k = kc * 32 + e + ((lane & 16) ? 16 : 0);
  int col = nt * 16 + (lane & 15);
  dst[idx] = (bhalf_t)W[k * Ncols + col];
}

// ---------------------------------------------------------------------------
// K1: zn = LN(z); a = zn@Wa+ba; b = zn@Wb+bb, written channel-major [c][i][k].
// A lane's 8 accumulator rows are 8 consecutive k -> single 16B store/matrix.
__global__ __launch_bounds__(256) void k1_ln_proj(
    const float* __restrict__ z, const float* __restrict__ zng,
    const float* __restrict__ znb, const bhalf_t* __restrict__ WaP,
    const float* __restrict__ ba, const bhalf_t* __restrict__ WbP,
    const float* __restrict__ bb, bhalf_t* __restrict__ a_t,
    bhalf_t* __restrict__ b_t) {
  __shared__ __attribute__((aligned(16))) float   zrow[16 * 132];
  __shared__ __attribute__((aligned(16))) bhalf_t zn[16 * 136];
  int t = threadIdx.x;
  size_t r0 = (size_t)blockIdx.x * 16;
  for (int fi = t; fi < 16 * 32; fi += 256) {  // coalesced b128 preload
    int row = fi >> 5, c4 = fi & 31;
    *(floatx4*)(zrow + row * 132 + c4 * 4) =
        *(const floatx4*)(z + (r0 + row) * CZ + c4 * 4);
  }
  __syncthreads();
  {  // layernorm 16 rows of Cz=128, 16 threads/row, from LDS
    int row = t >> 4, sub = t & 15;
    float sum = 0.f, ss = 0.f, vals[8];
    for (int k = 0; k < 8; k++) {
      float v = zrow[row * 132 + sub + 16 * k];
      vals[k] = v; sum += v; ss += v * v;
    }
    red16(sum, ss);
    float mean = sum * (1.f / CZ);
    float inv = rsqrtf(ss * (1.f / CZ) - mean * mean + 1e-5f);
    for (int k = 0; k < 8; k++) {
      int c = sub + 16 * k;
      zn[row * 136 + c] = (bhalf_t)((vals[k] - mean) * inv * zng[c] + znb[c]);
    }
  }
  __syncthreads();
  int wave = t >> 5, lane = t & 31;
  floatx8 acca = {}, accb = {};
  for (int kc = 0; kc < 4; kc++) {
    bhalf16 af = frag_a(zn, 136, kc * 32, lane);
    acca = wmma32(af, frag_pk(WaP, kc, wave, 8, lane), acca);
    accb = wmma32(af, frag_pk(WbP, kc, wave, 8, lane), accb);
  }
  int col = wave * 16 + (lane & 15);
  int hi = (lane & 16) ? 8 : 0;
  float biasa = ba[col], biasb = bb[col];
  int i = (int)(r0 / N_TOK);
  int kcol0 = (int)(r0 % N_TOK);
  bhalf8 pa, pb;
#pragma unroll
  for (int v = 0; v < 8; v++) {
    pa[v] = (bhalf_t)(acca[v] + biasa);
    pb[v] = (bhalf_t)(accb[v] + biasb);
  }
  size_t off = ((size_t)col * N_TOK + i) * N_TOK + (kcol0 + hi);
  *(bhalf8*)(a_t + off) = pa;   // one b128 store per matrix
  *(bhalf8*)(b_t + off) = pb;
}

// ---------------------------------------------------------------------------
// K2: tri[i,j,c] = (1/N) sum_k a[i,k,c]*b[j,k,c].  Per channel: A_c @ B_c^T.
// 2x2 register blocking per wave (32x32 output tile): 4 WMMA per 8 b128 loads.
// Grid channel-outer so a_t/b_t slices (1.18MB each) stay L2-resident.
__global__ __launch_bounds__(256) void k2_tri(const bhalf_t* __restrict__ a_t,
                                              const bhalf_t* __restrict__ b_t,
                                              bhalf_t* __restrict__ tri) {
  int b = blockIdx.x;                 // 128 ch * 12 iblk(64) * 6 jblk(128)
  int c = b / 72, rem = b % 72;
  int ib = rem / 6, jb = rem % 6;
  int wave = threadIdx.x >> 5, lane = threadIdx.x & 31;
  int wy = wave >> 2, wx = wave & 3;  // 2x4 wave grid -> 64x128 block tile
  int i0 = ib * 64 + wy * 32;
  int j0 = jb * 128 + wx * 32;
  const bhalf_t* A0 = a_t + ((size_t)c * N_TOK + i0) * N_TOK;
  const bhalf_t* A1 = A0 + 16 * N_TOK;
  const bhalf_t* B0 = b_t + ((size_t)c * N_TOK + j0) * N_TOK;
  const bhalf_t* B1 = B0 + 16 * N_TOK;
  floatx8 acc00 = {}, acc01 = {}, acc10 = {}, acc11 = {};
  int rowoff = (lane & 15) * N_TOK;
  for (int kb = 0; kb < 24; kb++) {
    if (kb < 23) {  // hide HBM/L2 latency on the next K-slab
      __builtin_prefetch(A0 + rowoff + (kb + 1) * 32, 0, 1);
      __builtin_prefetch(A1 + rowoff + (kb + 1) * 32, 0, 1);
      __builtin_prefetch(B0 + rowoff + (kb + 1) * 32, 0, 1);
      __builtin_prefetch(B1 + rowoff + (kb + 1) * 32, 0, 1);
    }
    bhalf16 a0 = frag_a(A0, N_TOK, kb * 32, lane);
    bhalf16 a1 = frag_a(A1, N_TOK, kb * 32, lane);
    bhalf16 b0 = frag_bt(B0, N_TOK, kb * 32, lane);
    bhalf16 b1 = frag_bt(B1, N_TOK, kb * 32, lane);
    acc00 = wmma32(a0, b0, acc00);
    acc01 = wmma32(a0, b1, acc01);
    acc10 = wmma32(a1, b0, acc10);
    acc11 = wmma32(a1, b1, acc11);
  }
  const float scale = 1.0f / N_TOK;
  int colL = lane & 15;
  int hi = (lane & 16) ? 8 : 0;
#pragma unroll
  for (int v = 0; v < 8; v++) {
    int m = v + hi;
    tri[((size_t)(i0 + m) * N_TOK + j0 + colL) * CZ + c] = (bhalf_t)(acc00[v] * scale);
    tri[((size_t)(i0 + m) * N_TOK + j0 + 16 + colL) * CZ + c] = (bhalf_t)(acc01[v] * scale);
    tri[((size_t)(i0 + 16 + m) * N_TOK + j0 + colL) * CZ + c] = (bhalf_t)(acc10[v] * scale);
    tri[((size_t)(i0 + 16 + m) * N_TOK + j0 + 16 + colL) * CZ + c] = (bhalf_t)(acc11[v] * scale);
  }
}

// ---------------------------------------------------------------------------
// K3: z1 = z + tri@Wout+bo;  h = silu(LN(z1)@W1+b1);  zout = z1 + h@W2+b2.
// tri tile staged in LDS once (not 8x per wave); residual in/out via
// coalesced b128; final store nontemporal (z stream is consumed by nothing).
// Also accumulates zsum[i][c] += sum_j zout (for mean over axis=1).
__global__ __launch_bounds__(256) void k3_zout_ffn(
    const bhalf_t* __restrict__ tri, const float* __restrict__ zin,
    const bhalf_t* __restrict__ WoP, const float* __restrict__ bo,
    const float* __restrict__ fg, const float* __restrict__ fb,
    const bhalf_t* __restrict__ W1P, const float* __restrict__ b1,
    const bhalf_t* __restrict__ W2P, const float* __restrict__ b2,
    float* __restrict__ zout, float* __restrict__ zsum) {
  __shared__ __attribute__((aligned(16))) float   znew[16 * 132];
  __shared__ __attribute__((aligned(16))) bhalf_t triL[16 * 136];
  __shared__ __attribute__((aligned(16))) bhalf_t hn[16 * 136];
  __shared__ __attribute__((aligned(16))) bhalf_t h1[16 * 264];
  int t = threadIdx.x;
  size_t r0 = (size_t)blockIdx.x * 16;
  for (int fi = t; fi < 16 * 32; fi += 256) {  // residual preload (b128)
    int row = fi >> 5, c4 = fi & 31;
    *(floatx4*)(znew + row * 132 + c4 * 4) =
        *(const floatx4*)(zin + (r0 + row) * CZ + c4 * 4);
  }
  {  // stage tri tile (16x128 bf16 = 4KB) once, coalesced b128
    int fi = t;  // 256 chunks of 8 bf16
    int row = fi >> 4, c8 = fi & 15;
    *(bhalf8*)(triL + row * 136 + c8 * 8) =
        *(const bhalf8*)(tri + (r0 + row) * CZ + c8 * 8);
  }
  __syncthreads();
  int wave = t >> 5, lane = t & 31;
  int colL = lane & 15, hi = (lane & 16) ? 8 : 0;
  {  // z_out projection (A from LDS)
    floatx8 acc = {};
    for (int kc = 0; kc < 4; kc++)
      acc = wmma32(frag_a(triL, 136, kc * 32, lane),
                   frag_pk(WoP, kc, wave, 8, lane), acc);
    int col = wave * 16 + colL;
    float bias = bo[col];
    for (int v = 0; v < 8; v++) znew[(v + hi) * 132 + col] += acc[v] + bias;
  }
  __syncthreads();
  {  // LN -> hn
    int row = t >> 4, sub = t & 15;
    float sum = 0.f, ss = 0.f, vals[8];
    for (int k = 0; k < 8; k++) {
      float v = znew[row * 132 + sub + 16 * k];
      vals[k] = v; sum += v; ss += v * v;
    }
    red16(sum, ss);
    float mean = sum * (1.f / CZ);
    float inv = rsqrtf(ss * (1.f / CZ) - mean * mean + 1e-5f);
    for (int k = 0; k < 8; k++) {
      int c = sub + 16 * k;
      hn[row * 136 + c] = (bhalf_t)((vals[k] - mean) * inv * fg[c] + fb[c]);
    }
  }
  __syncthreads();
  for (int tt = 0; tt < 2; tt++) {  // FFN1 128->256 + silu
    int nt = wave + tt * 8;
    floatx8 acc = {};
    for (int kc = 0; kc < 4; kc++)
      acc = wmma32(frag_a(hn, 136, kc * 32, lane),
                   frag_pk(W1P, kc, nt, 16, lane), acc);
    int col = nt * 16 + colL;
    float bias = b1[col];
    for (int v = 0; v < 8; v++) {
      float x = acc[v] + bias;
      h1[(v + hi) * 264 + col] = (bhalf_t)(x / (1.f + __expf(-x)));
    }
  }
  __syncthreads();
  {  // FFN2 256->128 + residual (in LDS) + column-sum for mean(z, axis=1)
    floatx8 acc = {};
    for (int kc = 0; kc < 8; kc++)
      acc = wmma32(frag_a(h1, 264, kc * 32, lane),
                   frag_pk(W2P, kc, wave, 8, lane), acc);
    int col = wave * 16 + colL;
    float bias = b2[col];
    float csum = 0.f;
    for (int v = 0; v < 8; v++) {
      float val = acc[v] + bias + znew[(v + hi) * 132 + col];
      znew[(v + hi) * 132 + col] = val;  // own element: no cross-lane hazard
      csum += val;
    }
    csum += __shfl_xor(csum, 16, 32);
    if (lane < 16) {
      int i = (int)(r0 / N_TOK);
      atomicAdd(&zsum[i * CZ + col], csum);
    }
  }
  __syncthreads();
  for (int fi = t; fi < 16 * 32; fi += 256) {  // coalesced NT b128 store
    int row = fi >> 5, c4 = fi & 31;
    floatx4 v4 = *(const floatx4*)(znew + row * 132 + c4 * 4);
    __builtin_nontemporal_store(v4, (floatx4*)(zout + (r0 + row) * CZ + c4 * 4));
  }
}

// ---------------------------------------------------------------------------
// K4: full s-track fused per 16-row block (48 blocks). ~62KB LDS.
__global__ __launch_bounds__(256) void k4_s_track(
    const float* __restrict__ s, const float* __restrict__ zsum,
    const float* __restrict__ sg, const float* __restrict__ sb,
    const bhalf_t* __restrict__ WsP, const float* __restrict__ bs,
    const bhalf_t* __restrict__ WzsP, const float* __restrict__ bzs,
    const float* __restrict__ fg, const float* __restrict__ fb,
    const bhalf_t* __restrict__ W1P, const float* __restrict__ b1,
    const bhalf_t* __restrict__ W2P, const float* __restrict__ b2,
    float* __restrict__ sout) {
  __shared__ __attribute__((aligned(16))) float   smid[16 * 388];
  __shared__ __attribute__((aligned(16))) bhalf_t hs[16 * 392];
  __shared__ __attribute__((aligned(16))) bhalf_t poolB[16 * 776];
  bhalf_t* zmb = poolB;             // [16][136], dead after P2
  bhalf_t* sn  = poolB + 16 * 136;  // [16][392], dead after P2
  bhalf_t* h1  = poolB;             // [16][776], lives P4..P5
  int t = threadIdx.x;
  int r0 = blockIdx.x * 16;
  for (int fi = t; fi < 16 * 96; fi += 256) {  // residual base = s (b128)
    int row = fi / 96, c4 = fi % 96;
    *(floatx4*)(smid + row * 388 + c4 * 4) =
        *(const floatx4*)(s + (size_t)(r0 + row) * CS + c4 * 4);
  }
  for (int idx = t; idx < 16 * CZ; idx += 256) {  // zmean = zsum / N
    int row = idx >> 7, col = idx & 127;
    zmb[row * 136 + col] = (bhalf_t)(zsum[(size_t)(r0 + row) * CZ + col] * (1.f / N_TOK));
  }
  __syncthreads();
  int wave = t >> 5, lane = t & 31;
  int colL = lane & 15, hi = (lane & 16) ? 8 : 0;
  {  // P1: LN(s) -> sn
    int row = t >> 4, sub = t & 15;
    float sum = 0.f, ss = 0.f, vals[24];
    for (int k = 0; k < 24; k++) {
      float v = smid[row * 388 + sub + 16 * k];
      vals[k] = v; sum += v; ss += v * v;
    }
    red16(sum, ss);
    float mean = sum * (1.f / CS);
    float inv = rsqrtf(ss * (1.f / CS) - mean * mean + 1e-5f);
    for (int k = 0; k < 24; k++) {
      int c = sub + 16 * k;
      sn[row * 392 + c] = (bhalf_t)((vals[k] - mean) * inv * sg[c] + sb[c]);
    }
  }
  __syncthreads();
  for (int tt = 0; tt < 3; tt++) {  // P2: smid += sn@Ws + zmean@Wzs + biases
    int nt = wave + tt * 8;
    floatx8 acc = {};
    for (int kc = 0; kc < 12; kc++)
      acc = wmma32(frag_a(sn, 392, kc * 32, lane), frag_pk(WsP, kc, nt, 24, lane), acc);
    for (int kc = 0; kc < 4; kc++)
      acc = wmma32(frag_a(zmb, 136, kc * 32, lane), frag_pk(WzsP, kc, nt, 24, lane), acc);
    int col = nt * 16 + colL;
    float bias = bs[col] + bzs[col];
    for (int v = 0; v < 8; v++) smid[(v + hi) * 388 + col] += acc[v] + bias;
  }
  __syncthreads();
  {  // P3: LN(smid) -> hs
    int row = t >> 4, sub = t & 15;
    float sum = 0.f, ss = 0.f, vals[24];
    for (int k = 0; k < 24; k++) {
      float v = smid[row * 388 + sub + 16 * k];
      vals[k] = v; sum += v; ss += v * v;
    }
    red16(sum, ss);
    float mean = sum * (1.f / CS);
    float inv = rsqrtf(ss * (1.f / CS) - mean * mean + 1e-5f);
    for (int k = 0; k < 24; k++) {
      int c = sub + 16 * k;
      hs[row * 392 + c] = (bhalf_t)((vals[k] - mean) * inv * fg[c] + fb[c]);
    }
  }
  __syncthreads();
  for (int tt = 0; tt < 6; tt++) {  // P4: FFN1 384->768 + silu
    int nt = wave + tt * 8;
    floatx8 acc = {};
    for (int kc = 0; kc < 12; kc++)
      acc = wmma32(frag_a(hs, 392, kc * 32, lane), frag_pk(W1P, kc, nt, 48, lane), acc);
    int col = nt * 16 + colL;
    float bias = b1[col];
    for (int v = 0; v < 8; v++) {
      float x = acc[v] + bias;
      h1[(v + hi) * 776 + col] = (bhalf_t)(x / (1.f + __expf(-x)));
    }
  }
  __syncthreads();
  for (int tt = 0; tt < 3; tt++) {  // P5: FFN2 768->384 + residual -> smid
    int nt = wave + tt * 8;
    floatx8 acc = {};
    for (int kc = 0; kc < 24; kc++)
      acc = wmma32(frag_a(h1, 776, kc * 32, lane), frag_pk(W2P, kc, nt, 24, lane), acc);
    int col = nt * 16 + colL;
    float bias = b2[col];
    for (int v = 0; v < 8; v++)
      smid[(v + hi) * 388 + col] += acc[v] + bias;  // own element
  }
  __syncthreads();
  for (int fi = t; fi < 16 * 96; fi += 256) {  // coalesced b128 store
    int row = fi / 96, c4 = fi % 96;
    *(floatx4*)(sout + (size_t)(r0 + row) * CS + c4 * 4) =
        *(const floatx4*)(smid + row * 388 + c4 * 4);
  }
}

// ---------------------------------------------------------------------------
extern "C" void kernel_launch(void* const* d_in, const int* in_sizes, int n_in,
                              void* d_out, int out_size, void* d_ws,
                              size_t ws_size, hipStream_t stream) {
  (void)in_sizes; (void)n_in; (void)out_size; (void)ws_size;
  const float* s    = (const float*)d_in[0];
  const float* z    = (const float*)d_in[1];
  const float* zn_g = (const float*)d_in[2];
  const float* zn_b = (const float*)d_in[3];
  const float* Wa   = (const float*)d_in[4];
  const float* ba   = (const float*)d_in[5];
  const float* Wb   = (const float*)d_in[6];
  const float* bb   = (const float*)d_in[7];
  const float* Wo   = (const float*)d_in[8];
  const float* bo   = (const float*)d_in[9];
  const float* sn_g = (const float*)d_in[10];
  const float* sn_b = (const float*)d_in[11];
  const float* Ws   = (const float*)d_in[12];
  const float* bs   = (const float*)d_in[13];
  const float* Wzs  = (const float*)d_in[14];
  const float* bzs  = (const float*)d_in[15];
  const float* fl_g = (const float*)d_in[16];
  const float* fl_b = (const float*)d_in[17];
  const float* Ws1  = (const float*)d_in[18];
  const float* bs1  = (const float*)d_in[19];
  const float* Ws2  = (const float*)d_in[20];
  const float* bs2  = (const float*)d_in[21];
  const float* zf_g = (const float*)d_in[22];
  const float* zf_b = (const float*)d_in[23];
  const float* Wz1  = (const float*)d_in[24];
  const float* bz1  = (const float*)d_in[25];
  const float* Wz2  = (const float*)d_in[26];
  const float* bz2  = (const float*)d_in[27];

  char* ws = (char*)d_ws;
  const size_t ZMAT = 150994944ull;  // 768*768*128 bf16 bytes
  bhalf_t* a_t  = (bhalf_t*)(ws);
  bhalf_t* b_t  = (bhalf_t*)(ws + ZMAT);
  bhalf_t* tri  = (bhalf_t*)(ws + 2 * ZMAT);
  float*   zsum = (float*)(ws + 3 * ZMAT);
  bhalf_t* wsp  = (bhalf_t*)(ws + 3 * ZMAT + 393216ull);
  bhalf_t* WaP  = wsp;
  bhalf_t* WbP  = WaP + 16384;
  bhalf_t* WoP  = WbP + 16384;
  bhalf_t* Wz1P = WoP + 16384;
  bhalf_t* Wz2P = Wz1P + 32768;
  bhalf_t* WsP  = Wz2P + 32768;
  bhalf_t* WzsP = WsP + 147456;
  bhalf_t* Ws1P = WzsP + 49152;
  bhalf_t* Ws2P = Ws1P + 294912;

  float* out_s = (float*)d_out;
  float* out_z = out_s + (size_t)N_TOK * CS;

  k_zero<<<384, 256, 0, stream>>>(zsum, N_TOK * CZ);
  k_pack_b<<<64, 256, 0, stream>>>(Wa, WaP, 128, 128);
  k_pack_b<<<64, 256, 0, stream>>>(Wb, WbP, 128, 128);
  k_pack_b<<<64, 256, 0, stream>>>(Wo, WoP, 128, 128);
  k_pack_b<<<128, 256, 0, stream>>>(Wz1, Wz1P, 128, 256);
  k_pack_b<<<128, 256, 0, stream>>>(Wz2, Wz2P, 256, 128);
  k_pack_b<<<576, 256, 0, stream>>>(Ws, WsP, 384, 384);
  k_pack_b<<<192, 256, 0, stream>>>(Wzs, WzsP, 128, 384);
  k_pack_b<<<1152, 256, 0, stream>>>(Ws1, Ws1P, 384, 768);
  k_pack_b<<<1152, 256, 0, stream>>>(Ws2, Ws2P, 768, 384);

  k1_ln_proj<<<36864, 256, 0, stream>>>(z, zn_g, zn_b, WaP, ba, WbP, bb, a_t, b_t);
  k2_tri<<<9216, 256, 0, stream>>>(a_t, b_t, tri);
  k3_zout_ffn<<<36864, 256, 0, stream>>>(tri, z, WoP, bo, zf_g, zf_b, Wz1P, bz1,
                                         Wz2P, bz2, out_z, zsum);
  k4_s_track<<<48, 256, 0, stream>>>(s, zsum, sn_g, sn_b, WsP, bs, WzsP, bzs,
                                     fl_g, fl_b, Ws1P, bs1, Ws2P, bs2, out_s);
}
